// NeuralModel_32959579029645
// MI455X (gfx1250) — matrix operand order repaired and checked by reading, not compiled, not run on hardware
//
#include <hip/hip_runtime.h>
#include <cmath>
#include <cstddef>
#include <cstdint>

// ---------------------------------------------------------------------------
// Model constants
// ---------------------------------------------------------------------------
#define FREQ_BINS 2049
#define NBANDS    64
#define EDIM      256
#define HIDDIM    682
#define HIDPAD    704                   // HIDDIM rounded to 64
#define T_LEN     256
#define NTOK      (NBANDS * T_LEN)      // 16384 tokens
#define L_AUD     261120
#define NHEAD     8
#define DHEAD     32                    // E / H

// Use CDNA5 async global->LDS DMA for f16 weight tiles (ASYNCcnt path).
#define USE_ASYNC_LDS 1

typedef __attribute__((ext_vector_type(16))) _Float16 v16h;
typedef __attribute__((ext_vector_type(8)))  float    v8f;
typedef __attribute__((ext_vector_type(8)))  _Float16 h8v;
typedef __attribute__((ext_vector_type(2)))  _Float16 h2v;

__device__ __forceinline__ float geluf(float x) {
  return 0.5f * x * (1.0f + erff(x * 0.70710678118654752f));
}
__device__ __forceinline__ float softplusf(float x) {
  return (x > 20.f) ? x : log1pf(expf(x));
}

__device__ __forceinline__ v8f wmma16(v16h a, v16h b, v8f c) {
  return __builtin_amdgcn_wmma_f32_16x16x32_f16(false, a, false, b, (short)0, c,
                                                false, false);
}

#if USE_ASYNC_LDS
// 16B per-lane async copy: global -> LDS, tracked by ASYNCcnt.
// LDS generic addresses keep the byte offset in the low 32 bits (ISA 10.2).
__device__ __forceinline__ void async_b128_to_lds(const void* gsrc, void* ldst) {
  unsigned l = (unsigned)(uintptr_t)ldst;
  asm volatile("global_load_async_to_lds_b128 %0, %1, off"
               :: "v"(l), "v"(gsrc) : "memory");
}
__device__ __forceinline__ void wait_async0() {
#if __has_builtin(__builtin_amdgcn_s_wait_asynccnt)
  __builtin_amdgcn_s_wait_asynccnt(0);
#else
  asm volatile("s_wait_asynccnt 0" ::: "memory");
#endif
}
#endif

// A fragment (16x32 f16) from LDS row-major [*, lda]; two contiguous 8-half
// runs per lane -> 2x ds_load_b128.  (lanes 0-15: K 0..7,16..23; 16-31: +8)
__device__ __forceinline__ v16h load_a_frag(const _Float16* A, int lda, int row0,
                                            int k0, int lane) {
  int g = lane >> 4, m = lane & 15;
  const _Float16* base = A + (row0 + m) * lda + k0 + 8 * g;
  v16h a;
  h8v* pa = reinterpret_cast<h8v*>(&a);
  pa[0] = *reinterpret_cast<const h8v*>(base);
  pa[1] = *reinterpret_cast<const h8v*>(base + 16);
  return a;
}

// B fragment (32x16 f16) from LDS in k-pair layout: B2[kp][n] = half2(k=2kp,2kp+1)
__device__ __forceinline__ v16h load_b_frag2(const h2v* B2, int ld2, int kp0,
                                             int n0, int lane) {
  int g = lane >> 4, n = lane & 15;
  const h2v* base = B2 + (kp0 + 4 * g) * ld2 + n0 + n;
  v16h b;
  h2v* pb = reinterpret_cast<h2v*>(&b);
#pragma unroll
  for (int i = 0; i < 8; i++) pb[i] = base[(i + (i & 4)) * ld2];
  return b;
}

// Pack f32 weight [K,N] -> f16 k-pair layout [Kp/2][Np] half2, zero padded.
__global__ __launch_bounds__(256) void k_packW(const float* __restrict__ W,
                                               _Float16* __restrict__ P,
                                               int K, int N, int Np, int total) {
  int i = blockIdx.x * 256 + threadIdx.x;
  if (i >= total) return;
  int kp = i / Np, n = i - kp * Np;
  float f0 = (2 * kp < K && n < N) ? W[(size_t)(2 * kp) * N + n] : 0.f;
  float f1 = (2 * kp + 1 < K && n < N) ? W[(size_t)(2 * kp + 1) * N + n] : 0.f;
  h2v v; v[0] = (_Float16)f0; v[1] = (_Float16)f1;
  reinterpret_cast<h2v*>(P)[i] = v;
}

// ---------------------------------------------------------------------------
// WMMA GEMM with packed f16 weights:
//   C[M,<Nvalid>] = act(A[M,K] @ Wpacked + bias) (+C residual)
// M%64==0, K%32==0, lda%4==0, B pre-padded to Np (%64==0).
// ---------------------------------------------------------------------------
__global__ __launch_bounds__(256) void k_gemmh(const float* __restrict__ A, int lda,
                                               const _Float16* __restrict__ P, int Np,
                                               const float* __restrict__ bias,
                                               float* __restrict__ C, int ldc,
                                               int M, int Nvalid, int K,
                                               int act, int addres) {
  __shared__ _Float16 As[64 * 32];
  __shared__ h2v Bs[16 * 64];
  const int tid = threadIdx.x, lane = tid & 31, wave = tid >> 5;
  const int m0 = blockIdx.y * 64, n0 = blockIdx.x * 64;
  const int r0 = (wave & 1) * 32, ct = wave >> 1;
  v8f acc0 = {}, acc1 = {};

  const int ar = tid >> 2, acb = (tid & 3) * 8;        // A: row, col-base
  const int br = tid >> 4, bo = (tid & 15) * 8;        // B: kp-row, half-off
  const float* srcA = A + (size_t)(m0 + ar) * lda + acb;
  const _Float16* srcB = P + (size_t)br * 2 * Np + (size_t)n0 * 2 + bo;
  _Float16* dstB = reinterpret_cast<_Float16*>(Bs) + br * 128 + bo;

  for (int kk = 0; kk < K; kk += 32) {
#if USE_ASYNC_LDS
    async_b128_to_lds(srcB + (size_t)kk * Np, dstB);
#else
    *reinterpret_cast<h8v*>(dstB) =
        *reinterpret_cast<const h8v*>(srcB + (size_t)kk * Np);
#endif
    {  // stage A (f32 -> f16), 8 contiguous elems/thread
      float4 f0 = *reinterpret_cast<const float4*>(srcA + kk);
      float4 f1 = *reinterpret_cast<const float4*>(srcA + kk + 4);
      h8v hv;
      hv[0] = (_Float16)f0.x; hv[1] = (_Float16)f0.y;
      hv[2] = (_Float16)f0.z; hv[3] = (_Float16)f0.w;
      hv[4] = (_Float16)f1.x; hv[5] = (_Float16)f1.y;
      hv[6] = (_Float16)f1.z; hv[7] = (_Float16)f1.w;
      *reinterpret_cast<h8v*>(&As[ar * 32 + acb]) = hv;
    }
    if (kk + 32 < K) {
      __builtin_prefetch(srcA + kk + 32, 0, 1);
      __builtin_prefetch(srcB + (size_t)(kk + 32) * Np, 0, 1);
    }
#if USE_ASYNC_LDS
    wait_async0();
#endif
    __syncthreads();
    v16h a0 = load_a_frag(As, 32, r0, 0, lane);
    v16h a1 = load_a_frag(As, 32, r0 + 16, 0, lane);
    v16h b  = load_b_frag2(Bs, 64, 0, ct * 16, lane);
    acc0 = wmma16(a0, b, acc0);
    acc1 = wmma16(a1, b, acc1);
    __syncthreads();
  }

  const int g = lane >> 4, n = lane & 15;
  const int col = n0 + ct * 16 + n;
  if (col < Nvalid) {
    float bv = bias ? bias[col] : 0.f;
    v8f accs[2] = {acc0, acc1};
#pragma unroll
    for (int t2 = 0; t2 < 2; t2++) {
#pragma unroll
      for (int r = 0; r < 8; r++) {
        int row = m0 + r0 + t2 * 16 + r + 8 * g;
        float v = accs[t2][r] + bv;
        if (act == 1) v = geluf(v);
        size_t idx = (size_t)row * ldc + col;
        if (addres) v += C[idx];
        C[idx] = v;
      }
    }
  }
}

// ---------------------------------------------------------------------------
// Fused gated FF (packed weights): C[M,HIDPAD] = gelu(A@GW+GB)*(A@UW+UB),
// cols >= HIDDIM written as 0 (zero-padded hidden for the down-projection).
// ---------------------------------------------------------------------------
__global__ __launch_bounds__(256) void k_gatedffh(const float* __restrict__ A, int lda,
                                                  const _Float16* __restrict__ PG,
                                                  const float* __restrict__ GB,
                                                  const _Float16* __restrict__ PU,
                                                  const float* __restrict__ UB,
                                                  float* __restrict__ C, int K) {
  __shared__ _Float16 As[64 * 32];
  __shared__ h2v Gs[16 * 64];
  __shared__ h2v Us[16 * 64];
  const int tid = threadIdx.x, lane = tid & 31, wave = tid >> 5;
  const int m0 = blockIdx.y * 64, n0 = blockIdx.x * 64;
  const int r0 = (wave & 1) * 32, ct = wave >> 1;
  v8f ag0 = {}, ag1 = {}, au0 = {}, au1 = {};

  const int ar = tid >> 2, acb = (tid & 3) * 8;
  const int br = tid >> 4, bo = (tid & 15) * 8;
  const float* srcA = A + (size_t)(m0 + ar) * lda + acb;
  const _Float16* srcG = PG + (size_t)br * 2 * HIDPAD + (size_t)n0 * 2 + bo;
  const _Float16* srcU = PU + (size_t)br * 2 * HIDPAD + (size_t)n0 * 2 + bo;
  _Float16* dstG = reinterpret_cast<_Float16*>(Gs) + br * 128 + bo;
  _Float16* dstU = reinterpret_cast<_Float16*>(Us) + br * 128 + bo;

  for (int kk = 0; kk < K; kk += 32) {
#if USE_ASYNC_LDS
    async_b128_to_lds(srcG + (size_t)kk * HIDPAD, dstG);
    async_b128_to_lds(srcU + (size_t)kk * HIDPAD, dstU);
#else
    *reinterpret_cast<h8v*>(dstG) =
        *reinterpret_cast<const h8v*>(srcG + (size_t)kk * HIDPAD);
    *reinterpret_cast<h8v*>(dstU) =
        *reinterpret_cast<const h8v*>(srcU + (size_t)kk * HIDPAD);
#endif
    {
      float4 f0 = *reinterpret_cast<const float4*>(srcA + kk);
      float4 f1 = *reinterpret_cast<const float4*>(srcA + kk + 4);
      h8v hv;
      hv[0] = (_Float16)f0.x; hv[1] = (_Float16)f0.y;
      hv[2] = (_Float16)f0.z; hv[3] = (_Float16)f0.w;
      hv[4] = (_Float16)f1.x; hv[5] = (_Float16)f1.y;
      hv[6] = (_Float16)f1.z; hv[7] = (_Float16)f1.w;
      *reinterpret_cast<h8v*>(&As[ar * 32 + acb]) = hv;
    }
    if (kk + 32 < K) {
      __builtin_prefetch(srcG + (size_t)(kk + 32) * HIDPAD, 0, 1);
      __builtin_prefetch(srcU + (size_t)(kk + 32) * HIDPAD, 0, 1);
    }
#if USE_ASYNC_LDS
    wait_async0();
#endif
    __syncthreads();
    v16h a0 = load_a_frag(As, 32, r0, 0, lane);
    v16h a1 = load_a_frag(As, 32, r0 + 16, 0, lane);
    v16h bg = load_b_frag2(Gs, 64, 0, ct * 16, lane);
    v16h bu = load_b_frag2(Us, 64, 0, ct * 16, lane);
    ag0 = wmma16(a0, bg, ag0);
    ag1 = wmma16(a1, bg, ag1);
    au0 = wmma16(a0, bu, au0);
    au1 = wmma16(a1, bu, au1);
    __syncthreads();
  }

  const int g = lane >> 4, n = lane & 15;
  const int col = n0 + ct * 16 + n;
  const bool ok = (col < HIDDIM);
  const float gb = ok ? GB[col] : 0.f, ub = ok ? UB[col] : 0.f;
  v8f gs[2] = {ag0, ag1};
  v8f us[2] = {au0, au1};
#pragma unroll
  for (int t2 = 0; t2 < 2; t2++) {
#pragma unroll
    for (int r = 0; r < 8; r++) {
      int row = m0 + r0 + t2 * 16 + r + 8 * g;
      float v = ok ? (geluf(gs[t2][r] + gb) * (us[t2][r] + ub)) : 0.f;
      C[(size_t)row * HIDPAD + col] = v;
    }
  }
}

// ---------------------------------------------------------------------------
// LayerNorm over E=256 per token
// ---------------------------------------------------------------------------
__global__ __launch_bounds__(256) void k_layernorm(const float* __restrict__ in,
                                                   const float* __restrict__ gw,
                                                   const float* __restrict__ bw,
                                                   float* __restrict__ out) {
  __shared__ float red[256];
  size_t t = blockIdx.x;
  int e = threadIdx.x;
  float x = in[t * 256 + e];
  red[e] = x; __syncthreads();
  for (int s = 128; s > 0; s >>= 1) { if (e < s) red[e] += red[e + s]; __syncthreads(); }
  float mean = red[0] / 256.f; __syncthreads();
  float d = x - mean;
  red[e] = d * d; __syncthreads();
  for (int s = 128; s > 0; s >>= 1) { if (e < s) red[e] += red[e + s]; __syncthreads(); }
  float var = red[0] / 256.f;
  out[t * 256 + e] = d * rsqrtf(var + 1e-5f) * gw[e] + bw[e];
}

// ---------------------------------------------------------------------------
// Fused PoPE attention, one block per (sequence, head). S in {64, 256}.
// ---------------------------------------------------------------------------
__global__ __launch_bounds__(256) void k_attention(const float* __restrict__ Q,
                                                   const float* __restrict__ Kin,
                                                   const float* __restrict__ V,
                                                   float* __restrict__ O,
                                                   const float* __restrict__ pbias,
                                                   int S, int sSeq, int sS) {
  __shared__ h2v keT2[32 * 256];      // [kp=d][t]: (cos,sin)*km pair
  __shared__ h2v vT2[128 * 32];       // [kp=s/2][d]: (V[s],V[s+1]) pair
  __shared__ _Float16 qe[32 * 64];    // [32 rows][2D]
  __shared__ float    sc[32 * 256];
  __shared__ _Float16 pr[32 * 256];

  const int tid = threadIdx.x, lane = tid & 31, wave = tid >> 5;
  const int seq = blockIdx.x, h = blockIdx.y;
  const float* pb = pbias + h * DHEAD;
  const float kln = 0.28782313662425572f;    // ln(10000)/32
  const float scale = 0.17677669529663687f;  // 1/sqrt(32)

  _Float16* vh = reinterpret_cast<_Float16*>(vT2);
  for (int idx = tid; idx < S * DHEAD; idx += 256) {
    int t = idx >> 5, d = idx & 31;
    size_t tok = (size_t)(seq * sSeq + t * sS);
    float km = softplusf(Kin[tok * 256 + h * DHEAD + d]);
    float invd = expf(-(float)d * kln);
    float bb = fminf(fmaxf(pb[d], -6.283185307179586f), 0.f);
    float ang = (float)t * invd + bb;
    h2v kp; kp[0] = (_Float16)(km * cosf(ang)); kp[1] = (_Float16)(km * sinf(ang));
    keT2[d * 256 + t] = kp;
    vh[(t >> 1) * 64 + d * 2 + (t & 1)] = (_Float16)V[tok * 256 + h * DHEAD + d];
  }
  __syncthreads();

  const int CT = S >> 4;
  for (int q0 = 0; q0 < S; q0 += 32) {
    for (int idx = tid; idx < 32 * DHEAD; idx += 256) {
      int r = idx >> 5, d = idx & 31;
      int s = q0 + r;
      size_t tok = (size_t)(seq * sSeq + s * sS);
      float qm = softplusf(Q[tok * 256 + h * DHEAD + d]);
      float invd = expf(-(float)d * kln);
      float ang = (float)s * invd;
      qe[r * 64 + 2 * d]     = (_Float16)(qm * cosf(ang));
      qe[r * 64 + 2 * d + 1] = (_Float16)(qm * sinf(ang));
    }
    __syncthreads();

    {  // scores[32,S] = qe @ keT  (K=2D=64)
      int rt = wave & 1;
      for (int ctc = (wave >> 1); ctc < CT; ctc += 4) {
        v8f acc = {};
#pragma unroll
        for (int k0 = 0; k0 < 64; k0 += 32) {
          v16h a = load_a_frag(qe, 64, rt * 16, k0, lane);
          v16h b = load_b_frag2(keT2, 256, k0 >> 1, ctc * 16, lane);
          acc = wmma16(a, b, acc);
        }
        int g = lane >> 4, n = lane & 15;
#pragma unroll
        for (int r = 0; r < 8; r++)
          sc[(rt * 16 + r + 8 * g) * 256 + ctc * 16 + n] = acc[r] * scale;
      }
    }
    __syncthreads();

    {  // softmax: 8 lanes per row, shfl-xor width 8
      int row = tid >> 3, l8 = tid & 7;
      float mx = -1e30f;
      for (int c = l8; c < S; c += 8) mx = fmaxf(mx, sc[row * 256 + c]);
      for (int off = 4; off > 0; off >>= 1) mx = fmaxf(mx, __shfl_xor(mx, off, 8));
      float sm = 0.f;
      for (int c = l8; c < S; c += 8) {
        float ev = expf(sc[row * 256 + c] - mx);
        sc[row * 256 + c] = ev;
        sm += ev;
      }
      for (int off = 4; off > 0; off >>= 1) sm += __shfl_xor(sm, off, 8);
      float rs = 1.f / sm;
      for (int c = l8; c < S; c += 8)
        pr[row * 256 + c] = (_Float16)(sc[row * 256 + c] * rs);
    }
    __syncthreads();

    if (wave < 4) {  // out[32,32] = pr @ V  (K=S)
      int rt = wave & 1, nt = wave >> 1;
      v8f acc = {};
      for (int k0 = 0; k0 < S; k0 += 32) {
        v16h a = load_a_frag(pr, 256, rt * 16, k0, lane);
        v16h b = load_b_frag2(vT2, 32, k0 >> 1, nt * 16, lane);
        acc = wmma16(a, b, acc);
      }
      int g = lane >> 4, n = lane & 15;
#pragma unroll
      for (int r = 0; r < 8; r++) {
        int s = q0 + rt * 16 + r + 8 * g;
        size_t tok = (size_t)(seq * sSeq + s * sS);
        O[tok * 256 + h * DHEAD + nt * 16 + n] = acc[r];
      }
    }
    __syncthreads();
  }
}

// ---------------------------------------------------------------------------
// Band split (per-band launch)
// ---------------------------------------------------------------------------
__global__ __launch_bounds__(256) void k_bandsplit(const float* __restrict__ re,
                                                   const float* __restrict__ im,
                                                   const float* __restrict__ W,
                                                   const float* __restrict__ b,
                                                   float* __restrict__ XS,
                                                   int s0, int bs) {
  __shared__ float xb[512];
  int t = blockIdx.x;
  int K = 4 * bs;
  for (int idx = threadIdx.x; idx < K; idx += 256) {
    int c = idx / bs, f = idx - c * bs;
    const float* src = (c < 2) ? (re + (size_t)c * FREQ_BINS * T_LEN)
                               : (im + (size_t)(c - 2) * FREQ_BINS * T_LEN);
    xb[idx] = src[(size_t)(s0 + f) * T_LEN + t];
  }
  __syncthreads();
  int e = threadIdx.x;
  float acc = b[e];
  for (int k = 0; k < K; k++) acc += xb[k] * W[k * 256 + e];
  XS[(size_t)t * 256 + e] = acc;
}

// conv1: [128,2,4096] stride 1024 pad 2048, + gelu (trimmed loop, no guards)
__global__ __launch_bounds__(256) void k_conv1(const float* __restrict__ a,
                                               const float* __restrict__ W,
                                               const float* __restrict__ b,
                                               float* __restrict__ out) {
  int oc = blockIdx.x, t = threadIdx.x;
  int base = t * 1024 - 2048;
  int k0 = base < 0 ? -base : 0;
  int k1 = L_AUD - base; if (k1 > 4096) k1 = 4096;
  float acc = b[oc];
  for (int ic = 0; ic < 2; ic++) {
    const float* ap = a + (size_t)ic * L_AUD + base;
    const float* wp = W + (size_t)oc * 8192 + ic * 4096;
    for (int k = k0; k < k1; k++) acc += ap[k] * wp[k];
  }
  out[oc * 256 + t] = geluf(acc);
}

// conv2: [256,128,3] pad 1 (+bias)
__global__ __launch_bounds__(256) void k_conv2(const float* __restrict__ c1,
                                               const float* __restrict__ W,
                                               const float* __restrict__ b,
                                               float* __restrict__ xa) {
  int oc = blockIdx.x, t = threadIdx.x;
  float acc = b[oc];
  for (int ic = 0; ic < 128; ic++) {
#pragma unroll
    for (int k = 0; k < 3; k++) {
      int p = t - 1 + k;
      if (p >= 0 && p < 256) acc += c1[ic * 256 + p] * W[oc * 384 + ic * 3 + k];
    }
  }
  xa[oc * 256 + t] = acc;
}

// concat: T1[token, 0:256]=XS, [256:512]=xa[:,t]
__global__ __launch_bounds__(256) void k_concat(const float* __restrict__ XS,
                                                const float* __restrict__ XA,
                                                float* __restrict__ T1) {
  size_t tok = blockIdx.x;
  int t = (int)(tok & 255), e = threadIdx.x;
  T1[tok * 512 + e]       = XS[tok * 256 + e];
  T1[tok * 512 + 256 + e] = XA[e * 256 + t];
}

// scatter (per-band): out[m, s0+f, t] = Y[t, 8*s0 + m*bs + f]
__global__ __launch_bounds__(256) void k_scatter(const float* __restrict__ Y,
                                                 float* __restrict__ out,
                                                 int s0, int bs) {
  int idx = blockIdx.x * 256 + threadIdx.x;
  int m = idx / (bs * 256);
  int rem = idx - m * bs * 256;
  int f = rem >> 8, t = rem & 255;
  out[((size_t)m * FREQ_BINS + s0 + f) * T_LEN + t] =
      Y[(size_t)t * (8 * FREQ_BINS) + 8 * s0 + m * bs + f];
}

// ---------------------------------------------------------------------------
// Host orchestration
// ---------------------------------------------------------------------------
// Flat input indices (setup_inputs dict insertion order, lists expanded):
// 0 real, 1 imag, 2 audio, 3..66 split_W, 67..130 split_b, 131 c1W, 132 c1b,
// 133 c2W, 134 c2b, 135 fusW, 136 fusb, 137 n0g, 138 n0b,
// 139..153 enc.time {n1g,n1b,wq,wk,wv,wo,pbias,n2g,n2b,gw,gb,uw,ub,dw,db},
// 154..168 enc.freq, 169 finalg, 170 finalb, 171..234 mergeW, 235..298 mergeb
extern "C" void kernel_launch(void* const* d_in, const int* in_sizes, int n_in,
                              void* d_out, int out_size, void* d_ws, size_t ws_size,
                              hipStream_t stream) {
  (void)in_sizes; (void)n_in; (void)out_size; (void)ws_size;
  const float* const* in = (const float* const*)d_in;
  float* out = (float*)d_out;
  float* ws = (float*)d_ws;

  // band edges (replicates make_band_splits; nearbyint == np.round half-even)
  int boff[64], bsz[64];
  {
    double ed[65];
    for (int i = 0; i <= 64; i++)
      ed[i] = std::nearbyint(std::pow((double)i / 64.0, 2.2) * 2049.0);
    ed[0] = 0.0; ed[64] = 2049.0;
    for (int i = 1; i <= 64; i++) if (ed[i] <= ed[i - 1]) ed[i] = ed[i - 1] + 1.0;
    int start = 0;
    for (int i = 0; i < 64; i++) {
      int end = (int)ed[i + 1];
      if (end > 2049) end = 2049;
      if (end <= start) end = (start + 1 < 2049) ? start + 1 : 2049;
      boff[i] = start; bsz[i] = end - start; start = end;
    }
    if (boff[63] + bsz[63] != 2049) bsz[63] = 2049 - boff[63];
  }

  // workspace layout (floats); ~151 MB total
  const size_t SZ_X = (size_t)NTOK * 256;
  float* X  = ws;
  float* T1 = X + SZ_X;                         // [NTOK, HIDPAD] / concat / Y
  float* Qb = T1 + (size_t)NTOK * HIDPAD;
  float* Kb = Qb + SZ_X;
  float* Vb = Kb + SZ_X;
  float* Ob = Vb + SZ_X;
  float* C1 = Ob + SZ_X;                        // [128,256]
  float* XA = C1 + 128 * 256;                   // [256,256]
  _Float16* PK = (_Float16*)(XA + 256 * 256);   // packed f16 weights
  size_t cur = 0;

  auto pack = [&](const float* W, int K, int N, int Kp, int Np) -> const _Float16* {
    _Float16* dst = PK + cur;
    int total = (Kp / 2) * Np;
    k_packW<<<(total + 255) / 256, 256, 0, stream>>>(W, dst, K, N, Np, total);
    cur += (size_t)total * 2;
    return dst;
  };

  // pack all GEMM weights once
  const _Float16* pFus = pack(in[135], 512, 256, 512, 256);
  const _Float16 *pWq[2][3], *pWk[2][3], *pWv[2][3], *pWo[2][3];
  const _Float16 *pGw[2][3], *pUw[2][3], *pDw[2][3];
  for (int p = 0; p < 2; p++) {
    int base = (p == 0) ? 139 : 154;
    for (int d = 0; d < 3; d++) {
      pWq[p][d] = pack(in[base + 2] + (size_t)d * EDIM * EDIM, 256, 256, 256, 256);
      pWk[p][d] = pack(in[base + 3] + (size_t)d * EDIM * EDIM, 256, 256, 256, 256);
      pWv[p][d] = pack(in[base + 4] + (size_t)d * EDIM * EDIM, 256, 256, 256, 256);
      pWo[p][d] = pack(in[base + 5] + (size_t)d * EDIM * EDIM, 256, 256, 256, 256);
      pGw[p][d] = pack(in[base + 9]  + (size_t)d * EDIM * HIDDIM, 256, HIDDIM, 256, HIDPAD);
      pUw[p][d] = pack(in[base + 11] + (size_t)d * EDIM * HIDDIM, 256, HIDDIM, 256, HIDPAD);
      pDw[p][d] = pack(in[base + 13] + (size_t)d * HIDDIM * EDIM, HIDDIM, 256, HIDPAD, 256);
    }
  }
  const _Float16* pMrg[64];
  int mrgNp[64];
  for (int bnd = 0; bnd < 64; bnd++) {
    int N = 8 * bsz[bnd];
    mrgNp[bnd] = (N + 63) & ~63;
    pMrg[bnd] = pack(in[171 + bnd], 256, N, 256, mrgNp[bnd]);
  }

  // 1) band split -> XS (stored in Qb region)
  for (int bnd = 0; bnd < 64; bnd++) {
    k_bandsplit<<<T_LEN, 256, 0, stream>>>(in[0], in[1], in[3 + bnd], in[67 + bnd],
                                           Qb + (size_t)bnd * T_LEN * 256,
                                           boff[bnd], bsz[bnd]);
  }
  // 2) audio path
  k_conv1<<<128, 256, 0, stream>>>(in[2], in[131], in[132], C1);
  k_conv2<<<256, 256, 0, stream>>>(C1, in[133], in[134], XA);
  // 3) fusion: concat -> gelu(GEMM) -> LN
  k_concat<<<NTOK, 256, 0, stream>>>(Qb, XA, T1);
  k_gemmh<<<dim3(4, NTOK / 64), 256, 0, stream>>>(T1, 512, pFus, 256, in[136],
                                                  X, 256, NTOK, 256, 512, 1, 0);
  k_layernorm<<<NTOK, 256, 0, stream>>>(X, in[137], in[138], X);

  // 4) dual-path encoder
  for (int d = 0; d < 3; d++) {
    for (int p = 0; p < 2; p++) {
      int base = (p == 0) ? 139 : 154;
      const float* n1g = in[base + 0] + d * EDIM;
      const float* n1b = in[base + 1] + d * EDIM;
      const float* pbp = in[base + 6] + (size_t)d * NHEAD * DHEAD;
      const float* n2g = in[base + 7] + d * EDIM;
      const float* n2b = in[base + 8] + d * EDIM;
      const float* gb  = in[base + 10] + (size_t)d * HIDDIM;
      const float* ub  = in[base + 12] + (size_t)d * HIDDIM;
      const float* db  = in[base + 14] + (size_t)d * EDIM;
      int S    = (p == 0) ? T_LEN : NBANDS;
      int nSeq = (p == 0) ? NBANDS : T_LEN;
      int sSeq = (p == 0) ? T_LEN : 1;
      int sS   = (p == 0) ? 1 : T_LEN;

      k_layernorm<<<NTOK, 256, 0, stream>>>(X, n1g, n1b, T1);
      k_gemmh<<<dim3(4, NTOK / 64), 256, 0, stream>>>(T1, 256, pWq[p][d], 256,
          nullptr, Qb, 256, NTOK, 256, 256, 0, 0);
      k_gemmh<<<dim3(4, NTOK / 64), 256, 0, stream>>>(T1, 256, pWk[p][d], 256,
          nullptr, Kb, 256, NTOK, 256, 256, 0, 0);
      k_gemmh<<<dim3(4, NTOK / 64), 256, 0, stream>>>(T1, 256, pWv[p][d], 256,
          nullptr, Vb, 256, NTOK, 256, 256, 0, 0);
      k_attention<<<dim3(nSeq, NHEAD), 256, 0, stream>>>(Qb, Kb, Vb, Ob, pbp,
                                                         S, sSeq, sS);
      // x += O @ wo
      k_gemmh<<<dim3(4, NTOK / 64), 256, 0, stream>>>(Ob, 256, pWo[p][d], 256,
          nullptr, X, 256, NTOK, 256, 256, 0, 1);
      // FF
      k_layernorm<<<NTOK, 256, 0, stream>>>(X, n2g, n2b, Ob);
      k_gatedffh<<<dim3(HIDPAD / 64, NTOK / 64), 256, 0, stream>>>(
          Ob, 256, pGw[p][d], gb, pUw[p][d], ub, T1, 256);
      k_gemmh<<<dim3(4, NTOK / 64), 256, 0, stream>>>(T1, HIDPAD, pDw[p][d], 256,
          db, X, 256, NTOK, 256, HIDPAD, 0, 1);
    }
  }

  // 5) final LN -> Ob
  k_layernorm<<<NTOK, 256, 0, stream>>>(X, in[169], in[170], Ob);

  // 6) band merge GEMMs into Y (=T1, [256 x 8*2049]) then scatter
  for (int bnd = 0; bnd < 64; bnd++) {
    int N = 8 * bsz[bnd];
    k_gemmh<<<dim3(mrgNp[bnd] / 64, 4), 256, 0, stream>>>(
        Ob + (size_t)bnd * T_LEN * 256, 256, pMrg[bnd], mrgNp[bnd],
        in[235 + bnd], T1 + 8 * boff[bnd], 8 * FREQ_BINS, T_LEN, N, 256, 0, 0);
  }
  for (int bnd = 0; bnd < 64; bnd++) {
    k_scatter<<<8 * bsz[bnd], 256, 0, stream>>>(T1, out, boff[bnd], bsz[bnd]);
  }
}
